// Generator_90555090469559
// MI455X (gfx1250) — matrix operand order repaired and checked by reading, not compiled, not run on hardware
//
#include <hip/hip_runtime.h>
#include <hip/hip_bf16.h>

#define N_NODES 16384
#define D_INF   64
#define D_HIDF  256
#define NEDGE   1048576

typedef __attribute__((ext_vector_type(16))) __bf16 v16bf;
typedef __attribute__((ext_vector_type(8)))  __bf16 v8bf;
typedef __attribute__((ext_vector_type(8)))  float  v8f;
typedef __attribute__((ext_vector_type(4)))  float  f4v;

__device__ __forceinline__ float leaky(float x){ return x >= 0.f ? x : 0.2f*x; }
__device__ __forceinline__ __bf16 to_bf16(float f){ return (__bf16)f; }
// order-preserving float->uint encoding for atomic max
__device__ __forceinline__ unsigned fenc(float f){
  unsigned u = __float_as_uint(f);
  return (u & 0x80000000u) ? ~u : (u | 0x80000000u);
}
__device__ __forceinline__ float fdec(unsigned u){
  return __uint_as_float((u & 0x80000000u) ? (u & 0x7fffffffu) : ~u);
}

// ---------------- init scratch that must start at zero ----------------
__global__ void k_init(float* __restrict__ outg, float* __restrict__ den,
                       unsigned* __restrict__ emax, float* __restrict__ colsum,
                       float* __restrict__ colsq){
  int i = blockIdx.x*blockDim.x + threadIdx.x;           // 1,048,576 threads
  outg[i] = 0.f;
  if (i < N_NODES){ den[i] = 0.f; emax[i] = 0u; }
  if (i < D_HIDF){ colsum[i] = 0.f; colsq[i] = 0.f; }
}

// ---------------- A-side prep: At[m][k] = bf16(x[k*64+m]) ----------------
__global__ void k_prep_at(const float* __restrict__ x, __bf16* __restrict__ At){
  int i = blockIdx.x*blockDim.x + threadIdx.x;           // 64*16384
  int m = i >> 14, k = i & 16383;
  At[i] = to_bf16(x[(size_t)k*64 + m]);
}

// ---------------- style-conditioned gamma/beta (6 matvecs) ----------------
// P layout: [0:64) g1 | [64:128) b1 | [128:384) g2 | [384:640) b2 | [640:704) g3 | [704:768) b3
__global__ void k_style(const float* __restrict__ style,
                        const float* a1w1, const float* a1b1, const float* a1w2, const float* a1b2,
                        const float* a2w1, const float* a2b1, const float* a2w2, const float* a2b2,
                        const float* a3w1, const float* a3b1, const float* a3w2, const float* a3b2,
                        float* __restrict__ P){
  __shared__ float s[64];
  if (threadIdx.x < 64) s[threadIdx.x] = style[threadIdx.x];
  __syncthreads();
  int t = blockIdx.x*blockDim.x + threadIdx.x;
  if (t >= 768) return;
  const float *W, *B; int row;
  if      (t < 64) { W=a1w1; B=a1b1; row=t;     }
  else if (t <128) { W=a1w2; B=a1b2; row=t-64;  }
  else if (t <384) { W=a2w1; B=a2b1; row=t-128; }
  else if (t <640) { W=a2w2; B=a2b2; row=t-384; }
  else if (t <704) { W=a3w1; B=a3b1; row=t-640; }
  else             { W=a3w2; B=a3b2; row=t-704; }
  float acc = B[row];
  const float* wr = W + row*64;
  #pragma unroll 8
  for (int j = 0; j < 64; j++) acc += wr[j]*s[j];
  P[t] = acc;
}

// ---------------- GEMM1 (split-K): h1[64,16384] = At(bf16) x trs_w^T ----------------
// block = 8 waves; block owns one 16-wide N tile; wave w covers K slice
// [w*2048, (w+1)*2048) -> 8192 waves feed the 23.3 TB/s trs_w stream.
// Scalar trip count; only the long-latency NT B-stream is software-pipelined
// (one iteration ahead, wrap instead of branch); A fragments are L2-hot and
// loaded inline. unroll(1) keeps live ranges small -> no scratch spills.
__global__ void k_gemm1(const __bf16* __restrict__ At, const float* __restrict__ trs_w,
                        float* __restrict__ h1){
  __shared__ float red[8*1024];                           // 32 KB
  int tid  = threadIdx.x;
  int wave = tid >> 5;                                    // 0..7 : K slice
  int lane = tid & 31;
  int lo = lane & 15, hi = lane >> 4;
  int nt = blockIdx.x;                                    // 0..1023 n tile
  int n  = nt*16 + lo;
  int k0 = wave * 2048;

  v8f acc[4];
  #pragma unroll
  for (int mt = 0; mt < 4; mt++)
    #pragma unroll
    for (int r = 0; r < 8; r++) acc[mt][r] = 0.f;

  const float* bbase = trs_w + (size_t)n*16384 + hi*16;   // B[k][n] = trs_w[n*16384+k]

  // prologue: B raw floats for it=0 (K = kb + hi*16 + e, 64B/lane, NT stream)
  f4v bq0, bq1, bq2, bq3;
  {
    const f4v* bp = (const f4v*)(bbase + k0);
    bq0 = __builtin_nontemporal_load(bp);
    bq1 = __builtin_nontemporal_load(bp+1);
    bq2 = __builtin_nontemporal_load(bp+2);
    bq3 = __builtin_nontemporal_load(bp+3);
  }

  #pragma unroll 1
  for (int it = 0; it < 64; ++it){
    int kb = k0 + it*32;
    // convert current B block to bf16 fragment
    v16bf bf;
    bf[0]=to_bf16(bq0.x);  bf[1]=to_bf16(bq0.y);  bf[2]=to_bf16(bq0.z);  bf[3]=to_bf16(bq0.w);
    bf[4]=to_bf16(bq1.x);  bf[5]=to_bf16(bq1.y);  bf[6]=to_bf16(bq1.z);  bf[7]=to_bf16(bq1.w);
    bf[8]=to_bf16(bq2.x);  bf[9]=to_bf16(bq2.y);  bf[10]=to_bf16(bq2.z); bf[11]=to_bf16(bq2.w);
    bf[12]=to_bf16(bq3.x); bf[13]=to_bf16(bq3.y); bf[14]=to_bf16(bq3.z); bf[15]=to_bf16(bq3.w);
    // issue next B block now (wraps on last iteration; harmless re-read)
    {
      int kbn = k0 + ((it + 1) & 63)*32;
      const f4v* bp = (const f4v*)(bbase + kbn);
      bq0 = __builtin_nontemporal_load(bp);
      bq1 = __builtin_nontemporal_load(bp+1);
      bq2 = __builtin_nontemporal_load(bp+2);
      bq3 = __builtin_nontemporal_load(bp+3);
    }
    // A fragments (L2-hot) + 4 WMMAs
    #pragma unroll
    for (int mt = 0; mt < 4; mt++){
      const __bf16* ap = At + (size_t)(mt*16 + lo)*16384 + kb;
      v8bf a0 = *(const v8bf*)(ap + hi*16);               // K = kb + hi*16 + e
      v8bf a1 = *(const v8bf*)(ap + hi*16 + 8);           // K = kb + hi*16 + 8 + e
      v16bf af;
      #pragma unroll
      for (int e = 0; e < 8; e++){ af[e] = a0[e]; af[8+e] = a1[e]; }
      acc[mt] = __builtin_amdgcn_wmma_f32_16x16x32_bf16(false, af, false, bf,
                                                        (short)0, acc[mt], false, false);
    }
  }

  // stash per-wave partials, then block-reduce (deterministic)
  #pragma unroll
  for (int mt = 0; mt < 4; mt++)
    #pragma unroll
    for (int r = 0; r < 8; r++)
      red[wave*1024 + (mt*8 + r)*32 + lane] = acc[mt][r];
  __syncthreads();
  #pragma unroll
  for (int j = 0; j < 4; j++){
    int e = tid + j*256;                                  // 0..1023 output elem of tile
    float s = 0.f;
    #pragma unroll
    for (int w = 0; w < 8; w++) s += red[w*1024 + e];
    int lc = e & 31, r = (e >> 5) & 7, mt = e >> 8;
    int row = mt*16 + (lc >> 4)*8 + r;
    int col = nt*16 + (lc & 15);
    h1[(size_t)row*16384 + col] = s;
  }
}

// ---------------- BN2 (over 64 rows) + leaky + transpose -> ht[16384,64] ----------------
__global__ void k_bn2t(const float* __restrict__ h1, const float* __restrict__ g,
                       const float* __restrict__ b, float* __restrict__ ht){
  int n = blockIdx.x*blockDim.x + threadIdx.x;            // 16384 columns
  float s = 0.f, s2 = 0.f;
  for (int d = 0; d < 64; d++){ float v = h1[(size_t)d*16384 + n]; s += v; s2 += v*v; }
  float m  = s * (1.f/64.f);
  float var = s2 * (1.f/64.f) - m*m;
  float rs = rsqrtf(var + 1e-5f);
  float gg = g[n]*rs, bb = b[n] - m*gg;
  for (int d = 0; d < 64; d++){
    float v = h1[(size_t)d*16384 + n];
    ht[(size_t)n*64 + d] = leaky(v*gg + bb);
  }
}

// ---------------- AdaIN over 64-feature rows (in place), wave per row ----------------
__global__ void k_adain64(float* __restrict__ p, const float* __restrict__ P,
                          int goff, int boff){
  int wid  = (blockIdx.x*blockDim.x + threadIdx.x) >> 5;
  int lane = threadIdx.x & 31;
  float* row = p + (size_t)wid*64;
  float v0 = row[lane], v1 = row[lane+32];
  float s = v0 + v1;
  #pragma unroll
  for (int m = 16; m; m >>= 1) s += __shfl_xor(s, m, 32);
  float mean = s * (1.f/64.f);
  float d0 = v0-mean, d1 = v1-mean;
  float ss = d0*d0 + d1*d1;
  #pragma unroll
  for (int m = 16; m; m >>= 1) ss += __shfl_xor(ss, m, 32);
  float inv = 1.f / (sqrtf(ss * (1.f/63.f)) + 1e-8f);     // ddof=1
  row[lane]    = P[goff+lane]   *d0*inv + P[boff+lane];
  row[lane+32] = P[goff+lane+32]*d1*inv + P[boff+lane+32];
}

// ---------------- GEMM2: h2[16384,256] = ht x fc1_w^T (fc1_b cancels in BN1) ----------------
__global__ void k_gemm2(const float* __restrict__ ht, const float* __restrict__ fc1_w,
                        float* __restrict__ h2){
  int wid  = (blockIdx.x*blockDim.x + threadIdx.x) >> 5;  // 16384 tiles
  int lane = threadIdx.x & 31;
  int lo = lane & 15, hi = lane >> 4;
  int mt = wid >> 4, nt = wid & 15;
  int m = mt*16 + lo, n = nt*16 + lo;
  v8f acc;
  #pragma unroll
  for (int r = 0; r < 8; r++) acc[r] = 0.f;
  #pragma unroll
  for (int kb = 0; kb < 64; kb += 32){
    v16bf af, bf;
    const float* ap = ht + (size_t)m*64 + kb;
    #pragma unroll
    for (int e = 0; e < 8; e++){ af[e] = to_bf16(ap[hi*16+e]); af[8+e] = to_bf16(ap[hi*16+8+e]); }
    const float* bp = fc1_w + (size_t)n*64 + kb + hi*16;  // B[k][n] = fc1_w[n*64+k]
    #pragma unroll
    for (int e = 0; e < 16; e++) bf[e] = to_bf16(bp[e]);
    acc = __builtin_amdgcn_wmma_f32_16x16x32_bf16(false, af, false, bf,
                                                  (short)0, acc, false, false);
  }
  #pragma unroll
  for (int r = 0; r < 8; r++)
    h2[(size_t)(mt*16 + hi*8 + r)*256 + nt*16 + lo] = acc[r];
}

// ---------------- BN1 stats (per column over 16384 rows), two stages ----------------
__global__ void k_bn1_part(const float* __restrict__ h2, float* __restrict__ colsum,
                           float* __restrict__ colsq){
  int c  = threadIdx.x;                                   // 256 columns
  int r0 = blockIdx.x*64;
  float s = 0.f, ss = 0.f;
  for (int r = 0; r < 64; r++){
    float v = h2[(size_t)(r0+r)*256 + c]; s += v; ss += v*v;
  }
  atomicAdd(&colsum[c], s); atomicAdd(&colsq[c], ss);
}
__global__ void k_bn1_fin(const float* __restrict__ colsum, const float* __restrict__ colsq,
                          const float* __restrict__ g, const float* __restrict__ b,
                          float* __restrict__ scale, float* __restrict__ shift){
  int c = threadIdx.x;
  float m  = colsum[c] * (1.f/16384.f);
  float var = colsq[c] * (1.f/16384.f) - m*m;
  float rs = rsqrtf(var + 1e-5f);
  scale[c] = g[c]*rs; shift[c] = b[c] - m*g[c]*rs;
}

// ---------------- BN1 apply + leaky + AdaIN2 over 256-feature rows ----------------
__global__ void k_bnadain256(float* __restrict__ h2, const float* __restrict__ scale,
                             const float* __restrict__ shift, const float* __restrict__ P){
  int wid  = (blockIdx.x*blockDim.x + threadIdx.x) >> 5;
  int lane = threadIdx.x & 31;
  float* row = h2 + (size_t)wid*256;
  float v[8]; float s = 0.f;
  #pragma unroll
  for (int j = 0; j < 8; j++){
    int c = lane + 32*j;
    float x = leaky(row[c]*scale[c] + shift[c]);
    v[j] = x; s += x;
  }
  #pragma unroll
  for (int m = 16; m; m >>= 1) s += __shfl_xor(s, m, 32);
  float mean = s * (1.f/256.f);
  float ss = 0.f;
  #pragma unroll
  for (int j = 0; j < 8; j++){ v[j] -= mean; ss += v[j]*v[j]; }
  #pragma unroll
  for (int m = 16; m; m >>= 1) ss += __shfl_xor(ss, m, 32);
  float inv = 1.f / (sqrtf(ss * (1.f/255.f)) + 1e-8f);
  #pragma unroll
  for (int j = 0; j < 8; j++){
    int c = lane + 32*j;
    row[c] = P[128+c]*v[j]*inv + P[384+c];
  }
}

// ---------------- GEMM3: xs[16384,64] = h2 x gat_w^T ----------------
__global__ void k_gemm3(const float* __restrict__ h2, const float* __restrict__ gat_w,
                        float* __restrict__ xs){
  int wid  = (blockIdx.x*blockDim.x + threadIdx.x) >> 5;  // 4096 tiles
  int lane = threadIdx.x & 31;
  int lo = lane & 15, hi = lane >> 4;
  int mt = wid >> 2, nt = wid & 3;
  int m = mt*16 + lo, n = nt*16 + lo;
  v8f acc;
  #pragma unroll
  for (int r = 0; r < 8; r++) acc[r] = 0.f;
  #pragma unroll
  for (int kb = 0; kb < 256; kb += 32){
    v16bf af, bf;
    const float* ap = h2 + (size_t)m*256 + kb;
    #pragma unroll
    for (int e = 0; e < 8; e++){ af[e] = to_bf16(ap[hi*16+e]); af[8+e] = to_bf16(ap[hi*16+8+e]); }
    const float* bp = gat_w + (size_t)n*256 + kb + hi*16; // B[k][n] = gat_w[n*256+k]
    #pragma unroll
    for (int e = 0; e < 16; e++) bf[e] = to_bf16(bp[e]);
    acc = __builtin_amdgcn_wmma_f32_16x16x32_bf16(false, af, false, bf,
                                                  (short)0, acc, false, false);
  }
  #pragma unroll
  for (int r = 0; r < 8; r++)
    xs[(size_t)(mt*16 + hi*8 + r)*64 + nt*16 + lo] = acc[r];
}

// ---------------- attention scores per node, wave per row ----------------
__global__ void k_att(const float* __restrict__ xs, const float* __restrict__ asv,
                      const float* __restrict__ adv, float* __restrict__ a_s,
                      float* __restrict__ a_d){
  int wid  = (blockIdx.x*blockDim.x + threadIdx.x) >> 5;
  int lane = threadIdx.x & 31;
  const float* row = xs + (size_t)wid*64;
  float v0 = row[lane], v1 = row[lane+32];
  float s = v0*asv[lane] + v1*asv[lane+32];
  float d = v0*adv[lane] + v1*adv[lane+32];
  #pragma unroll
  for (int m = 16; m; m >>= 1){ s += __shfl_xor(s, m, 32); d += __shfl_xor(d, m, 32); }
  if (lane == 0){ a_s[wid] = s; a_d[wid] = d; }
}

// ---------------- GAT edge passes ----------------
__global__ void k_edge1(const int* __restrict__ ei, const float* __restrict__ a_s,
                        const float* __restrict__ a_d, float* __restrict__ e_buf,
                        unsigned* __restrict__ emax){
  int i = blockIdx.x*blockDim.x + threadIdx.x;            // NEDGE threads
  int s = ei[i], d = ei[NEDGE + i];
  float e = leaky(a_s[s] + a_d[d]);
  e_buf[i] = e;
  atomicMax(&emax[d], fenc(e));
}
__global__ void k_edge2(const int* __restrict__ ei, float* __restrict__ e_buf,
                        const unsigned* __restrict__ emax, float* __restrict__ den){
  int i = blockIdx.x*blockDim.x + threadIdx.x;
  int d = ei[NEDGE + i];
  unsigned mu = emax[d];
  float m = (mu == 0u) ? 0.f : fdec(mu);                  // empty segment -> 0
  float ex = __expf(e_buf[i] - m);
  e_buf[i] = ex;
  atomicAdd(&den[d], ex);
}
__global__ void k_edge3(const int* __restrict__ ei, const float* __restrict__ e_buf,
                        const float* __restrict__ den, const float* __restrict__ xs,
                        float* __restrict__ outg){
  long long g = (long long)blockIdx.x*blockDim.x + threadIdx.x; // E*64 threads
  int i = (int)(g >> 6), f = (int)(g & 63);
  int s = ei[i], d = ei[NEDGE + i];
  float alpha = e_buf[i] / den[d];
  atomicAdd(&outg[(size_t)d*64 + f], alpha * xs[(size_t)s*64 + f]);
}

// ---------------- bias + leaky + AdaIN3 -> d_out ----------------
__global__ void k_final(const float* __restrict__ outg, const float* __restrict__ gbias,
                        const float* __restrict__ P, float* __restrict__ out){
  int wid  = (blockIdx.x*blockDim.x + threadIdx.x) >> 5;
  int lane = threadIdx.x & 31;
  const float* row = outg + (size_t)wid*64;
  float v0 = leaky(row[lane]    + gbias[lane]);
  float v1 = leaky(row[lane+32] + gbias[lane+32]);
  float s = v0 + v1;
  #pragma unroll
  for (int m = 16; m; m >>= 1) s += __shfl_xor(s, m, 32);
  float mean = s * (1.f/64.f);
  float d0 = v0-mean, d1 = v1-mean;
  float ss = d0*d0 + d1*d1;
  #pragma unroll
  for (int m = 16; m; m >>= 1) ss += __shfl_xor(ss, m, 32);
  float inv = 1.f / (sqrtf(ss * (1.f/63.f)) + 1e-8f);
  out[(size_t)wid*64 + lane]      = P[640+lane]   *d0*inv + P[704+lane];
  out[(size_t)wid*64 + lane + 32] = P[640+lane+32]*d1*inv + P[704+lane+32];
}

extern "C" void kernel_launch(void* const* d_in, const int* in_sizes, int n_in,
                              void* d_out, int out_size, void* d_ws, size_t ws_size,
                              hipStream_t stream){
  const float* x      = (const float*)d_in[0];
  const float* style  = (const float*)d_in[1];
  const int*   ei     = (const int*)  d_in[2];
  const float* trs_w  = (const float*)d_in[3];
  // d_in[4] trs_b : per-column constant, cancels in BN2 mean subtraction
  const float* bn2_g  = (const float*)d_in[5];
  const float* bn2_b  = (const float*)d_in[6];
  const float* fc1_w  = (const float*)d_in[7];
  // d_in[8] fc1_b : cancels in BN1
  const float* bn1_g  = (const float*)d_in[9];
  const float* bn1_b  = (const float*)d_in[10];
  const float* gat_w  = (const float*)d_in[11];
  const float* att_s  = (const float*)d_in[12];
  const float* att_d  = (const float*)d_in[13];
  const float* gat_b  = (const float*)d_in[14];
  const float* a1w1 = (const float*)d_in[15]; const float* a1b1 = (const float*)d_in[16];
  const float* a1w2 = (const float*)d_in[17]; const float* a1b2 = (const float*)d_in[18];
  const float* a2w1 = (const float*)d_in[19]; const float* a2b1 = (const float*)d_in[20];
  const float* a2w2 = (const float*)d_in[21]; const float* a2b2 = (const float*)d_in[22];
  const float* a3w1 = (const float*)d_in[23]; const float* a3b1 = (const float*)d_in[24];
  const float* a3w2 = (const float*)d_in[25]; const float* a3b2 = (const float*)d_in[26];

  char* ws = (char*)d_ws;
  const size_t MB = 1u << 20;
  __bf16*  At    = (__bf16*)ws;                 // 2 MB
  float*   h1    = (float*)(ws + 2*MB);         // 4 MB  [64,16384]
  float*   ht    = (float*)(ws + 6*MB);         // 4 MB  [16384,64]
  float*   h2    = (float*)(ws + 10*MB);        // 16 MB [16384,256]
  float*   xs    = (float*)(ws + 26*MB);        // 4 MB  [16384,64]
  float*   e_buf = (float*)(ws + 30*MB);        // 4 MB  [E]
  float*   outg  = (float*)(ws + 34*MB);        // 4 MB  [16384,64]
  char*    misc  = ws + 38*MB;
  float*    a_s    = (float*)misc;
  float*    a_d    = (float*)(misc + 65536);
  unsigned* emax   = (unsigned*)(misc + 2*65536);
  float*    den    = (float*)(misc + 3*65536);
  float*    colsum = (float*)(misc + 4*65536);
  float*    colsq  = (float*)(misc + 4*65536 + 1024);
  float*    bscale = (float*)(misc + 4*65536 + 2048);
  float*    bshift = (float*)(misc + 4*65536 + 3072);
  float*    P      = (float*)(misc + 4*65536 + 4096);   // 768 floats

  k_init   <<<4096, 256, 0, stream>>>(outg, den, emax, colsum, colsq);
  k_prep_at<<<4096, 256, 0, stream>>>(x, At);
  k_style  <<<3,    256, 0, stream>>>(style, a1w1,a1b1,a1w2,a1b2,
                                      a2w1,a2b1,a2w2,a2b2, a3w1,a3b1,a3w2,a3b2, P);
  k_gemm1  <<<1024, 256, 0, stream>>>(At, trs_w, h1);
  k_bn2t   <<<64,   256, 0, stream>>>(h1, bn2_g, bn2_b, ht);
  k_adain64<<<2048, 256, 0, stream>>>(ht, P, 0, 64);
  k_gemm2  <<<2048, 256, 0, stream>>>(ht, fc1_w, h2);
  k_bn1_part<<<256, 256, 0, stream>>>(h2, colsum, colsq);
  k_bn1_fin<<<1,    256, 0, stream>>>(colsum, colsq, bn1_g, bn1_b, bscale, bshift);
  k_bnadain256<<<2048, 256, 0, stream>>>(h2, bscale, bshift, P);
  k_gemm3  <<<512,  256, 0, stream>>>(h2, gat_w, xs);
  k_att    <<<2048, 256, 0, stream>>>(xs, att_s, att_d, a_s, a_d);
  k_edge1  <<<4096, 256, 0, stream>>>(ei, a_s, a_d, e_buf, emax);
  k_edge2  <<<4096, 256, 0, stream>>>(ei, e_buf, emax, den);
  k_edge3  <<<262144, 256, 0, stream>>>(ei, e_buf, den, xs, outg);
  k_final  <<<2048, 256, 0, stream>>>(outg, gat_b, P, (float*)d_out);

  (void)in_sizes; (void)n_in; (void)out_size; (void)ws_size;
}
